// DLSTM_83184926588953
// MI455X (gfx1250) — matrix operand order repaired and checked by reading, not compiled, not run on hardware
//
#include <hip/hip_runtime.h>

// ---------------------------------------------------------------------------
// CDNA5 (gfx1250) persistent recurrent-cell kernel.
// 4 independent workgroups x 16 batch rows, full T-loop per WG,
// bf16 WMMA (v_wmma_f32_16x16x32_bf16) with f32 accumulation,
// LDS-resident activations/carry, L2-resident bf16 weights.
// Weight pointers are laundered per time-step so the compiler cannot hoist
// the (t-invariant) weight loads out of the recurrence loop and spill them;
// all global accesses go through explicit addrspace(1) casts so they lower
// to global_load/global_store (not flat_*, which also ticks DScnt and ties
// up the LDS path).
// ---------------------------------------------------------------------------

typedef __attribute__((ext_vector_type(16))) __bf16 bf16x16;
typedef __attribute__((ext_vector_type(8)))  __bf16 bf16x8;
typedef __attribute__((ext_vector_type(8)))  float  f32x8;

// addrspace(1)-qualified views (pointer *into* global memory)
typedef bf16x16 __attribute__((address_space(1))) g_bf16x16;
typedef float   __attribute__((address_space(1))) g_float;

constexpr int kT      = 2048;
constexpr int kB      = 64;
constexpr int kI      = 256;
constexpr int kH      = 512;
constexpr int kZ      = kI + kH;   // 768
constexpr int k3H     = 3 * kH;    // 1536
constexpr int kRows   = 16;        // batch rows per workgroup (one WMMA M-tile)
constexpr int kWaves  = 16;
constexpr int kThreads = kWaves * 32;
constexpr float kEps  = 1e-5f;

__device__ __forceinline__ f32x8 vzero() { f32x8 z = {}; return z; }

__device__ __forceinline__ float eluf(float x) {
  return x > 0.f ? x : (__expf(x) - 1.f);
}

// Global loads/stores through addrspace(1): generic & global share address
// bits on AMDGCN, so the integer round-trip cast is an exact addrspacecast.
__device__ __forceinline__ float ldg_f32(const float* p) {
  return *(const g_float*)(unsigned long long)p;
}
__device__ __forceinline__ void stg_f32(float* p, float v) {
  *(g_float*)(unsigned long long)p = v;
}

// A-fragment (16x32 bf16): lanes 0-15 hold rows M=0..15 with
// K = kb..kb+15; lanes 16-31 hold the same
// rows with K = kb+16..kb+31.  -> two contiguous 16B LDS loads per lane.
__device__ __forceinline__ bf16x16 load_a_frag(const __bf16* base, int stride,
                                               int kb, int lane) {
  const int row = lane & 15;
  const int k0  = kb + ((lane >> 4) << 4);
  const __bf16* p = base + row * stride + k0;
  union { bf16x16 v; bf16x8 h[2]; } u;
  u.h[0] = *(const bf16x8*)(p);
  u.h[1] = *(const bf16x8*)(p + 8);
  return u.v;
}

// B-fragment (32x16 bf16): lane holds column N = nbase + (lane&15),
// K = kb + 16*(lane>>4) .. +15  -> one contiguous 32B global load per lane.
// W is stored [N][K] row-major (so out = act @ W^T needs no transpose).
__device__ __forceinline__ bf16x16 load_b_frag(const __bf16* W, int K,
                                               int nbase, int kb, int lane) {
  const int n  = nbase + (lane & 15);
  const int k0 = kb + ((lane >> 4) << 4);
  const __bf16* p = W + (size_t)n * K + k0;
  return *(const g_bf16x16*)(unsigned long long)p;   // global_load_b128 x2
}

__device__ __forceinline__ f32x8 wmma_bf16(bf16x16 a, bf16x16 b, f32x8 c) {
  // D = A*B + C : v_wmma_f32_16x16x32_bf16
  return __builtin_amdgcn_wmma_f32_16x16x32_bf16(
      /*neg_a=*/false, a, /*neg_b=*/false, b,
      /*c_mod=*/(short)0, c, /*reuse_a=*/false, /*reuse_b=*/false);
}

// Row-wise sum/sumsq of C-fragments into LDS stats arrays.
// C layout: lane L, element v -> row = v + 8*(L>>4), col = L&15.
// Butterfly over the 16 lanes of a half, then one ds_add_f32 per row.
__device__ __forceinline__ void tile_stats(const f32x8* acc, int ntiles,
                                           float* s_sum, float* s_sq, int lane) {
  for (int j = 0; j < ntiles; ++j) {
#pragma unroll
    for (int v = 0; v < 8; ++v) {
      float s = acc[j][v];
      float q = s * s;
#pragma unroll
      for (int m = 1; m < 16; m <<= 1) {
        s += __shfl_xor(s, m, 32);
        q += __shfl_xor(q, m, 32);
      }
      if ((lane & 15) == 0) {
        const int row = v + ((lane >> 4) << 3);
        atomicAdd(&s_sum[row], s);
        atomicAdd(&s_sq[row],  q);
      }
    }
  }
}

// One-shot f32 -> bf16 weight conversion into workspace (weights then live in L2).
__global__ void convert_weights(const float* __restrict__ W1,
                                const float* __restrict__ W2,
                                const float* __restrict__ W3,
                                __bf16* __restrict__ W1b,
                                __bf16* __restrict__ W2b,
                                __bf16* __restrict__ W3b) {
  const int i = blockIdx.x * blockDim.x + threadIdx.x;
  if (i < kH * kZ)  W1b[i] = (__bf16)W1[i];
  if (i < kH * kH)  W2b[i] = (__bf16)W2[i];
  if (i < k3H * kH) W3b[i] = (__bf16)W3[i];
}

__global__ __launch_bounds__(kThreads)
void dlstm_persistent(const float* __restrict__ x,
                      const __bf16* __restrict__ W1b,
                      const float* __restrict__ g1, const float* __restrict__ b1,
                      const __bf16* __restrict__ W2b,
                      const float* __restrict__ g2, const float* __restrict__ b2,
                      const __bf16* __restrict__ W3b,
                      const float* __restrict__ g3, const float* __restrict__ b3,
                      float* __restrict__ out) {
  extern __shared__ char smem[];
  __bf16* zbuf = (__bf16*)smem;                 // kRows x kZ   (z = [x | c])
  __bf16* h1   = zbuf + kRows * kZ;             // kRows x kH
  __bf16* h2   = h1   + kRows * kH;             // kRows x kH
  float*  cf   = (float*)(h2 + kRows * kH);     // kRows x kH   f32 carry
  float*  s0   = cf + kRows * kH;               // 2*kRows  (sum | sumsq)
  float*  s1   = s0 + 2 * kRows;
  float*  s2   = s1 + 2 * kRows;

  const int tid  = threadIdx.x;
  const int lane = tid & 31;
  const int wv   = tid >> 5;                    // wave id 0..15
  const int grp  = blockIdx.x;                  // owns rows [16*grp, 16*grp+16)
  const int rb   = (lane >> 4) << 3;            // row base of this lane's C half

  // init carry = 0 (f32 LDS + bf16 mirror inside zbuf) and stats
  for (int i = tid; i < kRows * kH; i += kThreads) {
    cf[i] = 0.f;
    zbuf[(i / kH) * kZ + kI + (i % kH)] = (__bf16)0.f;
  }
  if (tid < 2 * kRows) { s0[tid] = 0.f; s1[tid] = 0.f; s2[tid] = 0.f; }
  __syncthreads();

  const float invH  = 1.f / kH;
  const float inv3H = 1.f / k3H;

  // Mutable copies of the read-only pointers; laundered each time-step so the
  // weight loads cannot be hoisted out of the t-loop (and spilled to scratch).
  const __bf16* W1p = W1b;
  const __bf16* W2p = W2b;
  const __bf16* W3p = W3b;
  const float *g1p = g1, *b1p = b1, *g2p = g2, *b2p = b2, *g3p = g3, *b3p = b3;

  f32x8 acc[6];

  for (int t = 0; t < kT; ++t) {
    // opaque pointer laundering: new SSA values every iteration
    asm volatile("" : "+s"(W1p), "+s"(W2p), "+s"(W3p));
    asm volatile("" : "+s"(g1p), "+s"(b1p), "+s"(g2p),
                      "+s"(b2p), "+s"(g3p), "+s"(b3p));

    // ---- P0: stream x_t (f32 -> bf16) into z[:, 0:kI); prefetch x_{t+1}
    const float* xt = x + (size_t)t * kB * kI + (size_t)grp * kRows * kI;
    for (int i = tid; i < kRows * kI; i += kThreads) {
      zbuf[(i >> 8) * kZ + (i & (kI - 1))] = (__bf16)ldg_f32(xt + i);
    }
    if (t + 1 < kT) {
      __builtin_prefetch(x + (size_t)(t + 1) * kB * kI +
                         (size_t)grp * kRows * kI + tid * 8, 0, 1);
    }
    if (tid < 2 * kRows) s0[tid] = 0.f;
    __syncthreads();

    // ---- stage 1: z(16x768) @ W1^T -> 16x512 ; LN ; ELU -> h1
    acc[0] = vzero(); acc[1] = vzero();
    for (int kb = 0; kb < kZ; kb += 32) {
      bf16x16 a  = load_a_frag(zbuf, kZ, kb, lane);
      bf16x16 bL = load_b_frag(W1p, kZ, 16 * wv,        kb, lane);
      bf16x16 bH = load_b_frag(W1p, kZ, 16 * (wv + 16), kb, lane);
      acc[0] = wmma_bf16(a, bL, acc[0]);
      acc[1] = wmma_bf16(a, bH, acc[1]);
    }
    tile_stats(acc, 2, s0, s0 + kRows, lane);
    __syncthreads();
    {
      float mu[8], rs[8];
#pragma unroll
      for (int v = 0; v < 8; ++v) {
        const int r = rb + v;
        const float m = s0[r] * invH;
        mu[v] = m;
        rs[v] = rsqrtf(s0[kRows + r] * invH - m * m + kEps);
      }
#pragma unroll
      for (int j = 0; j < 2; ++j) {
        const int n = 16 * (wv + 16 * j) + (lane & 15);
        const float gm = ldg_f32(g1p + n), bt = ldg_f32(b1p + n);
#pragma unroll
        for (int v = 0; v < 8; ++v) {
          const float val = eluf((acc[j][v] - mu[v]) * rs[v] * gm + bt);
          h1[(rb + v) * kH + n] = (__bf16)val;
        }
      }
      if (tid < 2 * kRows) s1[tid] = 0.f;
    }
    __syncthreads();

    // ---- stage 2: h1(16x512) @ W2^T -> 16x512 ; LN ; ELU -> h2
    acc[0] = vzero(); acc[1] = vzero();
    for (int kb = 0; kb < kH; kb += 32) {
      bf16x16 a  = load_a_frag(h1, kH, kb, lane);
      bf16x16 bL = load_b_frag(W2p, kH, 16 * wv,        kb, lane);
      bf16x16 bH = load_b_frag(W2p, kH, 16 * (wv + 16), kb, lane);
      acc[0] = wmma_bf16(a, bL, acc[0]);
      acc[1] = wmma_bf16(a, bH, acc[1]);
    }
    tile_stats(acc, 2, s1, s1 + kRows, lane);
    __syncthreads();
    {
      float mu[8], rs[8];
#pragma unroll
      for (int v = 0; v < 8; ++v) {
        const int r = rb + v;
        const float m = s1[r] * invH;
        mu[v] = m;
        rs[v] = rsqrtf(s1[kRows + r] * invH - m * m + kEps);
      }
#pragma unroll
      for (int j = 0; j < 2; ++j) {
        const int n = 16 * (wv + 16 * j) + (lane & 15);
        const float gm = ldg_f32(g2p + n), bt = ldg_f32(b2p + n);
#pragma unroll
        for (int v = 0; v < 8; ++v) {
          const float val = eluf((acc[j][v] - mu[v]) * rs[v] * gm + bt);
          h2[(rb + v) * kH + n] = (__bf16)val;
        }
      }
      if (tid < 2 * kRows) s2[tid] = 0.f;
    }
    __syncthreads();

    // ---- stage 3: h2(16x512) @ W3^T -> 16x1536 ; LN over 1536 ; gates
    // wave wv owns column strips 16wv+256j for j=0..5:
    //   j=0,1 -> new_hx cols; j=2,3 -> new_cx (same cols +512); j=4,5 -> forget.
#pragma unroll
    for (int j = 0; j < 6; ++j) acc[j] = vzero();
    for (int kb = 0; kb < kH; kb += 32) {
      bf16x16 a = load_a_frag(h2, kH, kb, lane);
#pragma unroll
      for (int j = 0; j < 6; ++j) {
        bf16x16 b = load_b_frag(W3p, kH, 16 * (wv + 16 * j), kb, lane);
        acc[j] = wmma_bf16(a, b, acc[j]);
      }
    }
    tile_stats(acc, 6, s2, s2 + kRows, lane);
    __syncthreads();
    {
      float mu[8], rs[8];
#pragma unroll
      for (int v = 0; v < 8; ++v) {
        const int r = rb + v;
        const float m = s2[r] * inv3H;
        mu[v] = m;
        rs[v] = rsqrtf(s2[kRows + r] * inv3H - m * m + kEps);
      }
      // normalize in place
#pragma unroll
      for (int j = 0; j < 6; ++j) {
        const int n = 16 * (wv + 16 * j) + (lane & 15);
        const float gm = ldg_f32(g3p + n), bt = ldg_f32(b3p + n);
#pragma unroll
        for (int v = 0; v < 8; ++v) {
          acc[j][v] = (acc[j][v] - mu[v]) * rs[v] * gm + bt;
        }
      }
      // gating: f = sigmoid(forget)^2 ; c = f*elu(cx) + (1-f)*c ; out = elu(hx)
#pragma unroll
      for (int j = 0; j < 2; ++j) {
        const int col = 16 * wv + 256 * j + (lane & 15);   // 0..511
#pragma unroll
        for (int v = 0; v < 8; ++v) {
          const int r = rb + v;
          const float hx = eluf(acc[j][v]);
          const float cx = eluf(acc[j + 2][v]);
          const float sg = 1.f / (1.f + __expf(-acc[j + 4][v]));
          const float f  = sg * sg;
          const float cn = f * cx + (1.f - f) * cf[r * kH + col];
          cf[r * kH + col] = cn;
          zbuf[r * kZ + kI + col] = (__bf16)cn;
          stg_f32(out + (size_t)t * kB * kH +
                        (size_t)(grp * kRows + r) * kH + col, hx);
        }
      }
    }
    __syncthreads();
  }

  // final carry -> tail of d_out  (layout: outputs (T,B,H) then c_final (1,B,H))
  float* cfin = out + (size_t)kT * kB * kH + (size_t)grp * kRows * kH;
  for (int i = tid; i < kRows * kH; i += kThreads) stg_f32(cfin + i, cf[i]);
}

extern "C" void kernel_launch(void* const* d_in, const int* in_sizes, int n_in,
                              void* d_out, int out_size, void* d_ws, size_t ws_size,
                              hipStream_t stream) {
  (void)in_sizes; (void)n_in; (void)out_size; (void)ws_size;

  const float* x  = (const float*)d_in[0];
  const float* W1 = (const float*)d_in[1];
  const float* g1 = (const float*)d_in[2];
  const float* b1 = (const float*)d_in[3];
  const float* W2 = (const float*)d_in[4];
  const float* g2 = (const float*)d_in[5];
  const float* b2 = (const float*)d_in[6];
  const float* W3 = (const float*)d_in[7];
  const float* g3 = (const float*)d_in[8];
  const float* b3 = (const float*)d_in[9];
  float* out = (float*)d_out;

  __bf16* W1b = (__bf16*)d_ws;                       // 512x768
  __bf16* W2b = W1b + (size_t)kH * kZ;               // 512x512
  __bf16* W3b = W2b + (size_t)kH * kH;               // 1536x512

  const int nmax = k3H * kH;                         // 786432
  hipLaunchKernelGGL(convert_weights, dim3((nmax + 255) / 256), dim3(256), 0,
                     stream, W1, W2, W3, W1b, W2b, W3b);

  const size_t smem =
      (size_t)kRows * kZ * sizeof(__bf16) +          // zbuf
      2 * (size_t)kRows * kH * sizeof(__bf16) +      // h1, h2
      (size_t)kRows * kH * sizeof(float) +           // cf
      6 * kRows * sizeof(float);                     // s0,s1,s2

  hipLaunchKernelGGL(dlstm_persistent, dim3(kB / kRows), dim3(kThreads), smem,
                     stream, x, W1b, g1, b1, W2b, g2, b2, W3b, g3, b3, out);
}
